// BiAffineForward_61615600828572
// MI455X (gfx1250) — compile-verified, matrix operand order checked by reading, not run on hardware
//
#include <hip/hip_runtime.h>

typedef _Float16 v16h __attribute__((ext_vector_type(16)));
typedef _Float16 v8h  __attribute__((ext_vector_type(8)));
typedef float    v8f  __attribute__((ext_vector_type(8)));

#define NB   128
#define LL   512
#define DD   768
#define MM   24
#define HH   10
#define TTm  10
#define HT   100
#define FF   140
#define RR   97
#define RPAD 112   // 7*16 r-tiles
#define FPAD 160   // 5*32 k-blocks over FF
#define NPAD 144   // 9*16 n-tiles over FF
#define NP_ROWS (NB*HT)  // 12800
#define WROW 40    // padded halves per r-row in LDS W slice (bank spread, 16B aligned)

#if __has_builtin(__builtin_amdgcn_global_load_async_to_lds_b128) && \
    __has_builtin(__builtin_amdgcn_s_wait_asynccnt)
#define USE_ASYNC_LDS 1
#else
#define USE_ASYNC_LDS 0
#endif

#if USE_ASYNC_LDS
// Builtin signature (from hipcc diagnostic): param0 = int4 addrspace(1)*
// ("int __vector(4) __device__ *"), param1 = int4 addrspace(3)*, then 2 imm ints.
typedef int v4i32 __attribute__((vector_size(16)));
typedef __attribute__((address_space(1))) v4i32 g_v4i32;
typedef __attribute__((address_space(3))) v4i32 l_v4i32;
// Issue one decoupled 16B global->LDS transfer (tracked by ASYNCcnt).
// as(3) offset = low 32 bits of the shared-aperture flat address (ISA 10.2).
__device__ inline void stage_async_b128(const _Float16* gsrc, _Float16* ldst) {
  __builtin_amdgcn_global_load_async_to_lds_b128(
      (g_v4i32*)(unsigned long long)(const void*)gsrc,
      (l_v4i32*)(unsigned int)(unsigned long long)(void*)ldst, 0, 0);
}
#endif

// ---------------- span mean-pool -> f16 rows [N*M][768] ----------------
__global__ void k_span_mean(const float* __restrict__ sr, const int* __restrict__ spans,
                            _Float16* __restrict__ out) {
  int b = blockIdx.x;            // n*M + m
  int n = b / MM;
  int s = spans[b*2 + 0];
  int e = spans[b*2 + 1];
  float inv = 1.0f / (float)(e - s + 1);
  for (int idx = threadIdx.x; idx < DD; idx += blockDim.x) {
    float sum = 0.f;
    for (int j = s; j <= e; ++j) sum += sr[((size_t)n*LL + j)*DD + idx];
    out[(size_t)b*DD + idx] = (_Float16)(sum * inv);
  }
}

// ------------- transpose+pad weight f32[K][Nc] -> f16[Npad][Kpad] -------------
__global__ void k_wt(const float* __restrict__ W, _Float16* __restrict__ WT,
                     int kin, int kpad, int ncol) {
  int nIdx = blockIdx.x;
  for (int k = threadIdx.x; k < kpad; k += blockDim.x) {
    float v = (nIdx < ncol && k < kin) ? W[(size_t)k*ncol + nIdx] : 0.f;
    WT[(size_t)nIdx*kpad + k] = (_Float16)v;
  }
}

// lane fragment: two contiguous 8-half runs (16B each) per CDNA5 16-bit layout
__device__ inline v16h load_frag(const _Float16* p) {
  v8h lo = *(const v8h*)p;
  v8h hi = *(const v8h*)(p + 16);
  v16h r;
#pragma unroll
  for (int i = 0; i < 8; ++i) { r[i] = lo[i]; r[i+8] = hi[i]; }
  return r;
}

// ---------------- generic f16 WMMA GEMM: C[16][144] tile per wave ----------------
__global__ void __launch_bounds__(32) k_gemm_f16(
    const _Float16* __restrict__ A, const _Float16* __restrict__ BT,
    _Float16* __restrict__ C, int lda, int ldbt, int ldc, int kblocks, int relu) {
  int lane = threadIdx.x & 31;
  int n16 = lane & 15;
  int g   = lane >> 4;
  int row0 = blockIdx.x * 16;
  v8f acc[9];
#pragma unroll
  for (int i = 0; i < 9; ++i)
#pragma unroll
    for (int q = 0; q < 8; ++q) acc[i][q] = 0.f;

  const _Float16* arow = A + (size_t)(row0 + n16)*lda + g*8;
  const _Float16* brow = BT + (size_t)n16*ldbt + g*8;

  v16h af = load_frag(arow);
  for (int kb = 0; kb < kblocks; ++kb) {
    v16h af_next = af;
    if (kb + 1 < kblocks) af_next = load_frag(arow + (kb+1)*32);
    v16h bf = load_frag(brow + kb*32);
#pragma unroll
    for (int nt = 0; nt < 9; ++nt) {
      v16h bf_next = bf;
      if (nt < 8) bf_next = load_frag(brow + (size_t)(nt+1)*16*ldbt + kb*32);
      acc[nt] = __builtin_amdgcn_wmma_f32_16x16x32_f16(
          false, af, false, bf, (short)0, acc[nt], false, false);
      bf = bf_next;
    }
    af = af_next;
  }
#pragma unroll
  for (int nt = 0; nt < 9; ++nt)
#pragma unroll
    for (int v = 0; v < 8; ++v) {
      int m = v + 8*g;
      float x = acc[nt][v];
      if (relu) x = x > 0.f ? x : 0.f;
      C[(size_t)(row0 + m)*ldc + nt*16 + n16] = (_Float16)x;
    }
}

// ---------------- gather + mask -> hP/tP f16 [N*HT][FPAD] ----------------
__global__ void k_pack(const _Float16* __restrict__ mlph, const _Float16* __restrict__ mlpt,
                       const int* __restrict__ hmi, const int* __restrict__ hmm,
                       const int* __restrict__ tmi, const int* __restrict__ tmm,
                       const int* __restrict__ htc,
                       _Float16* __restrict__ hP, _Float16* __restrict__ tP) {
  int b = blockIdx.x;            // n*HT + p
  int n = b / HT;
  int hi = htc[b*2 + 0];
  int ti = htc[b*2 + 1];
  int hrow = n*HH  + hi;
  int trow = n*TTm + ti;
  int hmask = hmm[hrow];
  int tmask = tmm[trow];
  int hsrc = n*MM + hmi[hrow];
  int tsrc = n*MM + tmi[trow];
  for (int e = threadIdx.x; e < FPAD; e += blockDim.x) {
    _Float16 hv = (_Float16)0.f, tv = (_Float16)0.f;
    if (e < FF) {
      if (hmask) hv = mlph[(size_t)hsrc*FPAD + e];
      if (tmask) tv = mlpt[(size_t)tsrc*FPAD + e];
    }
    hP[(size_t)b*FPAD + e] = hv;
    tP[(size_t)b*FPAD + e] = tv;
  }
}

// ---------------- bili_W f32[R][FF][FF] -> padded f16 [RPAD][FF][FPAD] ----------------
__global__ void k_wconv(const float* __restrict__ W, _Float16* __restrict__ Wf) {
  int blk = blockIdx.x;          // r*FF + d over RPAD*FF
  int r = blk / FF, d = blk % FF;
  for (int e = threadIdx.x; e < FPAD; e += blockDim.x) {
    float v = (r < RR && e < FF) ? W[((size_t)r*FF + d)*FF + e] : 0.f;
    Wf[((size_t)r*FF + d)*FPAD + e] = (_Float16)v;
  }
}

// ---------------- bilinear: logits[p,r] = sum_{d,e} h[p,d] t[p,e] W[r,d,e] ----------------
// 4 waves/block share a double-buffered LDS slice of W (112 r x 32 e per (eb,d)).
// Staging uses GLOBAL_LOAD_ASYNC_TO_LDS_B128 (ASYNCcnt) so the DMA overlaps the
// 7 WMMAs; one s_wait_asynccnt + barrier per iteration. Register-staging fallback
// kept for toolchains without the async builtin.
__global__ void __launch_bounds__(128) k_bilinear(
    const _Float16* __restrict__ hP, const _Float16* __restrict__ tP,
    const _Float16* __restrict__ Wf, float* __restrict__ logits) {
  __shared__ alignas(16) _Float16 sW[2][RPAD*WROW];   // 2 * 8960 B

  int tid  = threadIdx.x;
  int lane = tid & 31;
  int wave = tid >> 5;
  int n16  = lane & 15;
  int g    = lane >> 4;
  int p0   = blockIdx.x*64 + wave*16;   // 16 p-rows per wave, 64 per block

  v8f acc[7];
#pragma unroll
  for (int i = 0; i < 7; ++i)
#pragma unroll
    for (int q = 0; q < 8; ++q) acc[i][q] = 0.f;

  const _Float16* trow = tP + (size_t)(p0 + n16)*FPAD + g*8;
  const _Float16* hrow = hP + (size_t)(p0 + n16)*FPAD;

  // per-thread staging chunk descriptors: 448 b128 chunks / 128 threads -> <=4 each
  bool cv[4];
  const _Float16* gbase[4];   // Wf + r*FF*FPAD + ch*8 ; add (d*FPAD + eb*32)
  _Float16* ld[2][4];         // LDS dst in each buffer
#pragma unroll
  for (int k = 0; k < 4; ++k) {
    int c = tid + k*128;
    cv[k] = c < 448;
    int r  = cv[k] ? (c >> 2) : 0;
    int ch = cv[k] ? (c & 3)  : 0;
    gbase[k] = Wf + (size_t)r*FF*FPAD + ch*8;
    ld[0][k] = &sW[0][r*WROW + ch*8];
    ld[1][k] = &sW[1][r*WROW + ch*8];
  }

#if !USE_ASYNC_LDS
  v8h sreg[4];
#endif

  // ---- prologue: stage (eb=0, d=0) into buffer 0
#if USE_ASYNC_LDS
#pragma unroll
  for (int k = 0; k < 4; ++k)
    if (cv[k]) stage_async_b128(gbase[k], ld[0][k]);
  __builtin_amdgcn_s_wait_asynccnt(0);
#else
#pragma unroll
  for (int k = 0; k < 4; ++k)
    if (cv[k]) sreg[k] = *(const v8h*)gbase[k];
#pragma unroll
  for (int k = 0; k < 4; ++k)
    if (cv[k]) *(v8h*)ld[0][k] = sreg[k];
#endif
  __syncthreads();

  int iter = 0;
  for (int eb = 0; eb < 5; ++eb) {
    v16h tf = load_frag(trow + eb*32);
    for (int d = 0; d < FF; ++d, ++iter) {
      int cur = iter & 1;
      int nxt = cur ^ 1;
      int carry = (d + 1 == FF) ? 1 : 0;
      int dn  = carry ? 0 : d + 1;
      int ebn = eb + carry;
      bool have_next = ebn < 5;
      size_t noff = (size_t)dn*FPAD + (size_t)ebn*32;

      // stage next slice into back buffer (async DMA overlaps WMMAs below)
      if (have_next) {
#if USE_ASYNC_LDS
#pragma unroll
        for (int k = 0; k < 4; ++k)
          if (cv[k]) stage_async_b128(gbase[k] + noff, ld[nxt][k]);
#else
#pragma unroll
        for (int k = 0; k < 4; ++k)
          if (cv[k]) sreg[k] = *(const v8h*)(gbase[k] + noff);
#endif
      }

      // compute on current buffer: A = h[p,d] * t[p, e-slice], B from LDS
      _Float16 h = hrow[d];
      v16h a;
#pragma unroll
      for (int i = 0; i < 16; ++i) a[i] = tf[i]*h;

#pragma unroll
      for (int rt = 0; rt < 7; ++rt) {
        const _Float16* lb = &sW[cur][(rt*16 + n16)*WROW + g*8];
        v8h lo = *(const v8h*)lb;
        v8h hi = *(const v8h*)(lb + 16);
        v16h bf;
#pragma unroll
        for (int i = 0; i < 8; ++i) { bf[i] = lo[i]; bf[i+8] = hi[i]; }
        acc[rt] = __builtin_amdgcn_wmma_f32_16x16x32_f16(
            false, a, false, bf, (short)0, acc[rt], false, false);
      }

      if (have_next) {
#if USE_ASYNC_LDS
        __builtin_amdgcn_s_wait_asynccnt(0);
#else
#pragma unroll
        for (int k = 0; k < 4; ++k)
          if (cv[k]) *(v8h*)ld[nxt][k] = sreg[k];
#endif
      }
      __syncthreads();
    }
  }

#pragma unroll
  for (int rt = 0; rt < 7; ++rt)
#pragma unroll
    for (int v = 0; v < 8; ++v)
      logits[(size_t)(p0 + v + 8*g)*RPAD + rt*16 + n16] = acc[rt][v];
}

// ---------------- masked logsumexp over p + bias ----------------
__global__ void k_lse(const float* __restrict__ logits, const int* __restrict__ mask,
                      const float* __restrict__ bias, float* __restrict__ out) {
  int n = blockIdx.x;
  int r = threadIdx.x;
  if (r >= RR) return;
  float mx = -3.0e38f;
  for (int p = 0; p < HT; ++p) {
    float x = logits[(size_t)(n*HT + p)*RPAD + r];
    if (mask[n*HT + p] == 0) x -= 1e10f;
    mx = fmaxf(mx, x);
  }
  float s = 0.f;
  for (int p = 0; p < HT; ++p) {
    float x = logits[(size_t)(n*HT + p)*RPAD + r];
    if (mask[n*HT + p] == 0) x -= 1e10f;
    s += __expf(x - mx);
  }
  out[(size_t)n*RR + r] = bias[r] + mx + __logf(s);
}

extern "C" void kernel_launch(void* const* d_in, const int* in_sizes, int n_in,
                              void* d_out, int out_size, void* d_ws, size_t ws_size,
                              hipStream_t stream) {
  (void)in_sizes; (void)n_in; (void)out_size; (void)ws_size;
  const float* sr   = (const float*)d_in[0];
  const int*   spans= (const int*)d_in[1];
  const int*   hmi  = (const int*)d_in[2];
  const int*   hmm  = (const int*)d_in[3];
  const int*   tmi  = (const int*)d_in[4];
  const int*   tmm  = (const int*)d_in[5];
  const int*   htc  = (const int*)d_in[6];
  const int*   htm  = (const int*)d_in[7];
  const float* W1h  = (const float*)d_in[8];
  const float* W2h  = (const float*)d_in[9];
  const float* W1t  = (const float*)d_in[10];
  const float* W2t  = (const float*)d_in[11];
  const float* bW   = (const float*)d_in[12];
  const float* bb   = (const float*)d_in[13];
  float* out = (float*)d_out;

  char* ws = (char*)d_ws;
  size_t off = 0;
  auto alloc = [&](size_t bytes) -> char* {
    char* p = ws + off;
    off = (off + bytes + 255) & ~((size_t)255);
    return p;
  };
  _Float16* spanemb = (_Float16*)alloc((size_t)NB*MM*DD*2);
  _Float16* W1hT    = (_Float16*)alloc((size_t)NPAD*DD*2);
  _Float16* W1tT    = (_Float16*)alloc((size_t)NPAD*DD*2);
  _Float16* W2hT    = (_Float16*)alloc((size_t)NPAD*FPAD*2);
  _Float16* W2tT    = (_Float16*)alloc((size_t)NPAD*FPAD*2);
  _Float16* midh    = (_Float16*)alloc((size_t)NB*MM*FPAD*2);
  _Float16* midt    = (_Float16*)alloc((size_t)NB*MM*FPAD*2);
  _Float16* mlph    = (_Float16*)alloc((size_t)NB*MM*FPAD*2);
  _Float16* mlpt    = (_Float16*)alloc((size_t)NB*MM*FPAD*2);
  _Float16* hP      = (_Float16*)alloc((size_t)NP_ROWS*FPAD*2);
  _Float16* tP      = (_Float16*)alloc((size_t)NP_ROWS*FPAD*2);
  _Float16* Wf      = (_Float16*)alloc((size_t)RPAD*FF*FPAD*2);
  float*    logitsP = (float*)   alloc((size_t)NP_ROWS*RPAD*4);

  // 1) span means (f16)
  k_span_mean<<<NB*MM, 256, 0, stream>>>(sr, spans, spanemb);

  // 2) weight transpose+pad to f16
  k_wt<<<NPAD, 256, 0, stream>>>(W1h, W1hT, DD, DD,   FF);
  k_wt<<<NPAD, 256, 0, stream>>>(W1t, W1tT, DD, DD,   FF);
  k_wt<<<NPAD, 256, 0, stream>>>(W2h, W2hT, FF, FPAD, FF);
  k_wt<<<NPAD, 256, 0, stream>>>(W2t, W2tT, FF, FPAD, FF);

  // 3) MLPs via WMMA; mid buffers zeroed so K-pad cols (144..159) are zero
  (void)hipMemsetAsync(midh, 0, (size_t)NB*MM*FPAD*2, stream);
  (void)hipMemsetAsync(midt, 0, (size_t)NB*MM*FPAD*2, stream);
  k_gemm_f16<<<(NB*MM)/16, 32, 0, stream>>>(spanemb, W1hT, midh, DD,   DD,   FPAD, DD/32,   1);
  k_gemm_f16<<<(NB*MM)/16, 32, 0, stream>>>(spanemb, W1tT, midt, DD,   DD,   FPAD, DD/32,   1);
  k_gemm_f16<<<(NB*MM)/16, 32, 0, stream>>>(midh,    W2hT, mlph, FPAD, FPAD, FPAD, FPAD/32, 0);
  k_gemm_f16<<<(NB*MM)/16, 32, 0, stream>>>(midt,    W2tT, mlpt, FPAD, FPAD, FPAD, FPAD/32, 0);

  // 4) gather+mask h/t pairs ; convert bilinear weights
  k_pack<<<NP_ROWS, 160, 0, stream>>>(mlph, mlpt, hmi, hmm, tmi, tmm, htc, hP, tP);
  k_wconv<<<RPAD*FF, 160, 0, stream>>>(bW, Wf);

  // 5) big bilinear on WMMA (48.7 GFLOP), async-DMA double-buffered W stream
  k_bilinear<<<NP_ROWS/64, 128, 0, stream>>>(hP, tP, Wf, logitsP);

  // 6) masked logsumexp + bias
  k_lse<<<NB, 128, 0, stream>>>(logitsP, htm, bb, out);
}
